// Model_83141976916076
// MI455X (gfx1250) — matrix operand order repaired and checked by reading, not compiled, not run on hardware
//
#include <hip/hip_runtime.h>

#define S_LEN 1024
#define BATCH 32
#define EMB   128
#define HEADS 4
#define HDIM  32
#define QK_SCALE 0.1767766952966369f   // 1/sqrt(32)
#define LN_EPS   1e-5f

typedef __bf16 bf16_t;
typedef __attribute__((ext_vector_type(16))) __bf16        v16bf;
typedef __attribute__((ext_vector_type(8)))  float          v8f;
typedef __attribute__((ext_vector_type(4)))  unsigned int   v4u;
typedef __attribute__((ext_vector_type(8)))  int            v8i;
typedef __attribute__((ext_vector_type(4)))  int            v4i;

// ---- WMMA fragment loaders (wave32, 16x16x32 bf16) ----
// A (16xK=32): lane m=lane&15; k = hi*8 + {0..7} and 16 + hi*8 + {0..7}
static __device__ __forceinline__ v16bf fragA_bf(const bf16_t* p, int stride, int lane) {
  int m = lane & 15, h8 = (lane >> 4) << 3;
  const bf16_t* r = p + m * stride;
  v16bf u;
#pragma unroll
  for (int j = 0; j < 8; ++j) { u[j] = r[h8 + j]; u[8 + j] = r[16 + h8 + j]; }
  return u;
}
static __device__ __forceinline__ v16bf fragA_f32(const float* p, int stride, int lane) {
  int m = lane & 15, h8 = (lane >> 4) << 3;
  const float* r = p + m * stride;
  v16bf u;
#pragma unroll
  for (int j = 0; j < 8; ++j) { u[j] = (bf16_t)r[h8 + j]; u[8 + j] = (bf16_t)r[16 + h8 + j]; }
  return u;
}
// B (K=32 x 16): source is N-major (B^T row-major): lane n=lane&15; k = hi*8 + {0..7} and 16 + hi*8 + {0..7}
static __device__ __forceinline__ v16bf fragB_bf(const bf16_t* p, int stride, int lane) {
  int n = lane & 15, k0 = (lane >> 4) << 3;
  const bf16_t* r = p + n * stride + k0;
  v16bf u;
#pragma unroll
  for (int j = 0; j < 8; ++j) { u[j] = r[j]; u[8 + j] = r[16 + j]; }
  return u;
}
static __device__ __forceinline__ v16bf fragB_f32(const float* p, int stride, int lane) {
  int n = lane & 15, k0 = (lane >> 4) << 3;
  const float* r = p + n * stride + k0;
  v16bf u;
#pragma unroll
  for (int j = 0; j < 8; ++j) { u[j] = (bf16_t)r[j]; u[8 + j] = (bf16_t)r[16 + j]; }
  return u;
}
static __device__ __forceinline__ v8f wmma_bf16(v16bf a, v16bf b, v8f c) {
  return __builtin_amdgcn_wmma_f32_16x16x32_bf16(false, a, false, b, (short)0, c, false, false);
}

// ---- Tensor Data Mover: 2-D tile (bf16) global -> LDS ----
// D# group0: [1:0]=count=1, [63:32]=lds_addr, [120:64]=global_addr, [127:126]=type=2
// D# group1: [17:16]=data_size(1=2B), [79:48]=tensor_dim0, [111:80]=tensor_dim1,
//            [127:112]=tile_dim0, [143:128]=tile_dim1, [207:160]=tensor_dim0_stride
static __device__ __forceinline__ void tdm_load_2d_bf16(
    unsigned lds_off, const void* gptr,
    unsigned tensor_d0, unsigned tensor_d1,
    unsigned tile_d0, unsigned tile_d1, unsigned stride0)
{
  unsigned long long ga = (unsigned long long)(size_t)gptr;
  v4u g0;
  g0[0] = 1u;                                        // count=1, user mode
  g0[1] = lds_off;                                   // lds_addr (bytes)
  g0[2] = (unsigned)ga;                              // global_addr[31:0]
  g0[3] = (unsigned)((ga >> 32) & 0x01FFFFFFu) | 0x80000000u;  // addr[56:32] | type=2
  v8i g1;
  g1[0] = 0x00010000;                                // data_size = 2 bytes
  g1[1] = (int)(tensor_d0 << 16);                    // tensor_dim0[15:0] at bit48
  g1[2] = (int)((tensor_d0 >> 16) | (tensor_d1 << 16));
  g1[3] = (int)((tensor_d1 >> 16) | (tile_d0 << 16));
  g1[4] = (int)tile_d1;                              // tile_dim1 (tile_dim2=0)
  g1[5] = (int)stride0;                              // tensor_dim0_stride[31:0]
  g1[6] = 0;
  g1[7] = 0;                                         // dim1_stride unused (2-D)
  v4i z4 = {0, 0, 0, 0};
  v8i z8 = {0, 0, 0, 0, 0, 0, 0, 0};
  __builtin_amdgcn_tensor_load_to_lds(g0, g1, z4, z4, z8, 0);
}

// =====================================================================
// Kernel 1: QKV projection.  qkv[t][f] = x[t]·W[f] + b[f]
// Writes: Q,K as (B,H,S,D) bf16 (Q pre-scaled), V^T as (B,H,D,S) bf16.
// =====================================================================
__global__ __launch_bounds__(256) void qkv_proj_kernel(
    const float* __restrict__ x, const float* __restrict__ w,
    const float* __restrict__ bias,
    bf16_t* __restrict__ qb, bf16_t* __restrict__ kb, bf16_t* __restrict__ vtb)
{
  const int tid = threadIdx.x, lane = tid & 31, wv = tid >> 5;
  const int tblk = blockIdx.x * 32;
#pragma unroll
  for (int i = 0; i < 6; ++i) {
    int tile = wv * 6 + i;                 // 48 tiles / 8 waves
    int tr = tile / 24, fc = tile % 24;
    int t0 = tblk + tr * 16, f0 = fc * 16;
    v8f acc = {};
#pragma unroll
    for (int ks = 0; ks < 4; ++ks) {       // K = 128 = 4 x 32
      v16bf a  = fragA_f32(x + (size_t)t0 * EMB + ks * 32, EMB, lane);
      v16bf bm = fragB_f32(w + (size_t)f0 * EMB + ks * 32, EMB, lane);
      acc = wmma_bf16(a, bm, acc);
    }
    int n = lane & 15, hi = lane >> 4;
    int f = f0 + n;
    float bv = bias[f];
    if (f0 < 128) {                        // Q (wave-uniform branch)
      int h = f >> 5, d = f & 31;
#pragma unroll
      for (int r = 0; r < 8; ++r) {
        int t = t0 + r + hi * 8;
        int s = t >> 5, b = t & 31;
        qb[(((size_t)(b * HEADS + h)) * S_LEN + s) * HDIM + d] =
            (bf16_t)((acc[r] + bv) * QK_SCALE);
      }
    } else if (f0 < 256) {                 // K
      int fk = f - 128, h = fk >> 5, d = fk & 31;
#pragma unroll
      for (int r = 0; r < 8; ++r) {
        int t = t0 + r + hi * 8;
        int s = t >> 5, b = t & 31;
        kb[(((size_t)(b * HEADS + h)) * S_LEN + s) * HDIM + d] = (bf16_t)(acc[r] + bv);
      }
    } else {                               // V^T
      int fv = f - 256, h = fv >> 5, d = fv & 31;
#pragma unroll
      for (int r = 0; r < 8; ++r) {
        int t = t0 + r + hi * 8;
        int s = t >> 5, b = t & 31;
        vtb[(((size_t)(b * HEADS + h)) * HDIM + d) * S_LEN + s] = (bf16_t)(acc[r] + bv);
      }
    }
  }
}

// =====================================================================
// Kernel 2: attention for one (b,h), 32 query rows per block.
// Full 32x1024 fp32 score tile lives in LDS; K/V tiles arrive via TDM.
// =====================================================================
__global__ __launch_bounds__(256) void attn_kernel(
    const bf16_t* __restrict__ qb, const bf16_t* __restrict__ kb,
    const bf16_t* __restrict__ vtb, bf16_t* __restrict__ attnb)
{
  __shared__ float  sS[32 * 1024];     // 128 KB score/prob tile
  __shared__ bf16_t sQ[32 * 32];       // 2 KB
  __shared__ bf16_t sTile[32 * 256];   // 16 KB (K chunk / V^T chunk)
  __shared__ float  sO[32 * 32];       // 4 KB split-K partials
  __shared__ float  sRed[32 * 8];
  __shared__ float  sSum[32];
  __shared__ float  sMax[32];

  const int tid = threadIdx.x, lane = tid & 31, wv = tid >> 5;
  const int qt = blockIdx.x, bh = blockIdx.y;
  const int b = bh >> 2, h = bh & 3;
  const size_t headoff = (size_t)bh * S_LEN * HDIM;

  const unsigned ldsQ = (unsigned)(size_t)(void*)sQ;
  const unsigned ldsT = (unsigned)(size_t)(void*)sTile;

  // Q tile 32x32 bf16 (contiguous): TDM as one 1024-element row
  if (wv == 0) {
    tdm_load_2d_bf16(ldsQ, qb + headoff + (size_t)qt * 32 * HDIM,
                     1024, 1, 1024, 1, 1024);
    __builtin_amdgcn_s_wait_tensorcnt(0);
  }
  __syncthreads();

  // ---- scores = Q K^T (D=32 -> single WMMA per 16x16 tile) ----
  const int rs = (wv >> 2) * 16;               // wave row base (2 x 4 wave grid)
  const v16bf qA = fragA_bf(sQ + rs * HDIM, HDIM, lane);
  const int cb = (wv & 3) * 32;                // wave col base within 128-key chunk

  for (int ch = 0; ch < 8; ++ch) {
    __syncthreads();
    if (wv == 0) {                             // K chunk: 128 rows x 32 elems
      tdm_load_2d_bf16(ldsT, kb + headoff + (size_t)ch * 128 * HDIM,
                       32, 1024, 32, 128, 32);
      __builtin_amdgcn_s_wait_tensorcnt(0);
    }
    __syncthreads();
#pragma unroll
    for (int ct = 0; ct < 2; ++ct) {
      int c0 = cb + ct * 16;
      v16bf kB = fragB_bf(sTile + c0 * HDIM, HDIM, lane);
      v8f acc = {};
      acc = wmma_bf16(qA, kB, acc);
      int n = lane & 15, hi = lane >> 4;
#pragma unroll
      for (int r = 0; r < 8; ++r)
        sS[(rs + r + hi * 8) * 1024 + ch * 128 + c0 + n] = acc[r];
    }
  }
  __syncthreads();

  // ---- softmax over rows (unnormalized exp; sum kept per row) ----
  {
    int row = tid >> 3, seg = tid & 7;         // 8 threads / row, 128 elems each
    float* p = sS + row * 1024 + seg * 128;
    float m = -3.0e38f;
    for (int i = 0; i < 128; ++i) m = fmaxf(m, p[i]);
    sRed[row * 8 + seg] = m;
    __syncthreads();
    if (seg == 0) {
      float mm = sRed[row * 8];
      for (int i = 1; i < 8; ++i) mm = fmaxf(mm, sRed[row * 8 + i]);
      sMax[row] = mm;
    }
    __syncthreads();
    float mm = sMax[row], sum = 0.f;
    for (int i = 0; i < 128; ++i) { float e = __expf(p[i] - mm); p[i] = e; sum += e; }
    sRed[row * 8 + seg] = sum;
    __syncthreads();
    if (seg == 0) {
      float ss = 0.f;
      for (int i = 0; i < 8; ++i) ss += sRed[row * 8 + i];
      sSum[row] = ss;
    }
  }

  // ---- O = P V  (split-K over 8 waves: 4 tiles x 2 k-halves) ----
  const int tO = wv & 3;
  const int ro = (tO >> 1) * 16, dbase = (tO & 1) * 16;
  const int khalf = wv >> 2;
  v8f o = {};
  for (int c = 0; c < 4; ++c) {                // 4 chunks of 256 keys
    __syncthreads();
    if (wv == 0) {                             // V^T chunk: 32 rows x 256 elems
      tdm_load_2d_bf16(ldsT, vtb + headoff + (size_t)c * 256,
                       1024, 32, 256, 32, 1024);
      __builtin_amdgcn_s_wait_tensorcnt(0);
    }
    __syncthreads();
#pragma unroll
    for (int ks = 0; ks < 4; ++ks) {
      int kk = khalf * 4 + ks;                 // wave's k-steps within chunk
      v16bf pA = fragA_f32(sS + ro * 1024 + c * 256 + kk * 32, 1024, lane);
      v16bf vB = fragB_bf(sTile + dbase * 256 + kk * 32, 256, lane);
      o = wmma_bf16(pA, vB, o);
    }
  }
  __syncthreads();
  int n = lane & 15, hi = lane >> 4;
  if (wv >= 4) {                               // stash high-half partials
#pragma unroll
    for (int r = 0; r < 8; ++r)
      sO[(ro + r + hi * 8) * 32 + dbase + n] = o[r];
  }
  __syncthreads();
  if (wv < 4) {                                // combine, normalize, store bf16
#pragma unroll
    for (int r = 0; r < 8; ++r) {
      int rl = ro + r + hi * 8;
      float val = (o[r] + sO[rl * 32 + dbase + n]) / sSum[rl];
      int s = qt * 32 + rl;
      attnb[((size_t)s * BATCH + b) * EMB + h * HDIM + dbase + n] = (bf16_t)val;
    }
  }
}

// =====================================================================
// Kernel 3: out-proj GEMM + bias + residual + LayerNorm. 16 tokens/block.
// =====================================================================
__global__ __launch_bounds__(256) void outproj_ln_kernel(
    const bf16_t* __restrict__ attnb, const float* __restrict__ wo,
    const float* __restrict__ bo, const float* __restrict__ x,
    const float* __restrict__ gamma, const float* __restrict__ beta,
    float* __restrict__ out)
{
  __shared__ bf16_t sW[128 * 128];     // 32 KB Wo in bf16
  __shared__ float sY[16 * 128];       // 8 KB pre-LN rows
  __shared__ float sRedA[16 * 16];
  __shared__ float sRedB[16 * 16];
  __shared__ float sMean[16];
  __shared__ float sRstd[16];

  const int tid = threadIdx.x, lane = tid & 31, wv = tid >> 5;
  const int t0 = blockIdx.x * 16;

#pragma unroll
  for (int i = 0; i < 64; ++i) sW[tid + i * 256] = (bf16_t)wo[tid + i * 256];
  __syncthreads();

  const int f0 = wv * 16;                      // 8 waves cover 128 features
  v8f acc = {};
#pragma unroll
  for (int ks = 0; ks < 4; ++ks) {
    v16bf a  = fragA_bf(attnb + (size_t)t0 * EMB + ks * 32, EMB, lane);
    v16bf bm = fragB_bf(sW + f0 * EMB + ks * 32, EMB, lane);
    acc = wmma_bf16(a, bm, acc);
  }
  int n = lane & 15, hi = lane >> 4;
  int f = f0 + n;
  float bv = bo[f];
#pragma unroll
  for (int r = 0; r < 8; ++r) {
    int m = r + hi * 8;
    sY[m * 128 + f] = acc[r] + bv + x[(size_t)(t0 + m) * EMB + f];
  }
  __syncthreads();

  int row = tid >> 4, seg = tid & 15;          // 16 threads / token row
  float* y = sY + row * 128 + seg * 8;
  float sm = 0.f, sq = 0.f;
#pragma unroll
  for (int i = 0; i < 8; ++i) { sm += y[i]; sq += y[i] * y[i]; }
  sRedA[row * 16 + seg] = sm; sRedB[row * 16 + seg] = sq;
  __syncthreads();
  if (seg == 0) {
    float sa = 0.f, sb = 0.f;
    for (int i = 0; i < 16; ++i) { sa += sRedA[row * 16 + i]; sb += sRedB[row * 16 + i]; }
    float mean = sa * (1.0f / 128.0f);
    float var  = sb * (1.0f / 128.0f) - mean * mean;
    sMean[row] = mean;
    sRstd[row] = rsqrtf(var + LN_EPS);
  }
  __syncthreads();
  float mean = sMean[row], rstd = sRstd[row];
#pragma unroll
  for (int i = 0; i < 8; ++i) {
    int fc = seg * 8 + i;
    out[(size_t)(t0 + row) * EMB + fc] = (y[i] - mean) * rstd * gamma[fc] + beta[fc];
  }
}

// =====================================================================
extern "C" void kernel_launch(void* const* d_in, const int* in_sizes, int n_in,
                              void* d_out, int out_size, void* d_ws, size_t ws_size,
                              hipStream_t stream)
{
  (void)in_sizes; (void)n_in; (void)out_size; (void)ws_size;
  const float* x  = (const float*)d_in[0];
  const float* wi = (const float*)d_in[1];
  const float* bi = (const float*)d_in[2];
  const float* wo = (const float*)d_in[3];
  const float* bo = (const float*)d_in[4];
  const float* g  = (const float*)d_in[5];
  const float* be = (const float*)d_in[6];
  float* out = (float*)d_out;

  const size_t HEAD_ELEMS = (size_t)BATCH * HEADS * S_LEN * HDIM;  // 4M
  bf16_t* qb    = (bf16_t*)d_ws;
  bf16_t* kb    = qb  + HEAD_ELEMS;
  bf16_t* vtb   = kb  + HEAD_ELEMS;
  bf16_t* attnb = vtb + HEAD_ELEMS;     // 32 MB total workspace

  qkv_proj_kernel  <<<dim3(1024),    256, 0, stream>>>(x, wi, bi, qb, kb, vtb);
  attn_kernel      <<<dim3(32, 128), 256, 0, stream>>>(qb, kb, vtb, attnb);
  outproj_ln_kernel<<<dim3(2048),    256, 0, stream>>>(attnb, wo, bo, x, g, be, out);
}
